// CausalSelfAttention_23931557773748
// MI455X (gfx1250) — compile-verified, matrix operand order checked by reading, not run on hardware
//
#include <hip/hip_runtime.h>
#include <hip/hip_bf16.h>

// ---------------------------------------------------------------------------
// CausalSelfAttention for MI455X (gfx1250, wave32, WMMA bf16 path)
// B=2, T=2048, D=2048, H=16, HD=128, KVH=4, GROUP=4
// ---------------------------------------------------------------------------

#define BATCH 2
#define SEQ   2048
#define DIM   2048
#define NH    16
#define NKVH  4
#define HDIM  128
#define QKVN  3072     // 2048 (q) + 512 (k) + 512 (v)
#define ROWS  (BATCH * SEQ)   // 4096

typedef __attribute__((ext_vector_type(16))) __bf16 v16bf;
typedef __attribute__((ext_vector_type(8)))  float  v8f;

union Frag {
    v16bf v;
    unsigned int u[8];
};

// 16-bit A/B fragment k-offset for element-pair i (0..7), half = lane>>4.
__device__ __forceinline__ int k0map(int i, int half) {
    return ((i & 3) << 1) + ((i >> 2) << 4) + (half << 3);
}

__device__ __forceinline__ v8f wmma_bf16(const v16bf& a, const v16bf& b, const v8f& c) {
    return __builtin_amdgcn_wmma_f32_16x16x32_bf16(false, a, false, b, (short)0, c, false, false);
}

__device__ __forceinline__ unsigned pack2(unsigned short lo, unsigned short hi) {
    return (unsigned)lo | ((unsigned)hi << 16);
}

// CDNA5 async DMA: global -> LDS, 16B per lane, tracked by ASYNCcnt.
// vdst = per-lane LDS byte address, vaddr = per-lane 64-bit global address.
__device__ __forceinline__ void async_b128_to_lds(unsigned lds_addr, const void* gptr) {
    asm volatile("global_load_async_to_lds_b128 %0, %1, off"
                 :: "v"(lds_addr), "v"(gptr) : "memory");
}
__device__ __forceinline__ void wait_asynccnt0() {
    asm volatile("s_wait_asynccnt 0x0" ::: "memory");
}
__device__ __forceinline__ unsigned lds_addr_of(const void* p) {
    return (unsigned)(unsigned long long)p;   // flat LDS aperture: addr[31:0] = LDS offset
}

// ---------------------------------------------------------------------------
// Elementwise converts
// ---------------------------------------------------------------------------
__global__ void cvt_f32_to_bf16(const float* __restrict__ src,
                                __bf16* __restrict__ dst, int n) {
    for (int i = blockIdx.x * blockDim.x + threadIdx.x; i < n;
         i += gridDim.x * blockDim.x) {
        dst[i] = (__bf16)src[i];
    }
}

__global__ void pack_wqkv_bf16(const float* __restrict__ Wq,
                               const float* __restrict__ Wk,
                               const float* __restrict__ Wv,
                               __bf16* __restrict__ W) {
    const int n = DIM * QKVN;
    for (int i = blockIdx.x * blockDim.x + threadIdx.x; i < n;
         i += gridDim.x * blockDim.x) {
        int r = i / QKVN;
        int c = i - r * QKVN;
        float v;
        if (c < 2048)       v = Wq[r * 2048 + c];
        else if (c < 2560)  v = Wk[r * 512 + (c - 2048)];
        else                v = Wv[r * 512 + (c - 2560)];
        W[i] = (__bf16)v;
    }
}

// ---------------------------------------------------------------------------
// Tiled WMMA GEMM: C[M,N] (f32) = A[M,K] (bf16 RM) * B[K,N] (bf16 RM)
// block = 256 threads (8 waves), tile 128x128, K-step 32, double-buffered LDS.
// A tiles staged by async global->LDS DMA (ASYNCcnt); B tiles transposed
// through registers. Wave grid 4(M) x 2(N): 8 WMMAs per wave per step.
// ---------------------------------------------------------------------------
__global__ void gemm_bf16_f32(const __bf16* __restrict__ A,
                              const __bf16* __restrict__ B,
                              float* __restrict__ C,
                              int M, int N, int K) {
    __shared__ __bf16 As[2][128][32];   // row-major A tiles
    __shared__ __bf16 Bs[2][128][36];   // transposed B tiles (col-major), padded

    const int nblk = blockIdx.x * 128;
    const int mblk = blockIdx.y * 128;
    const int wave = threadIdx.x >> 5;
    const int lane = threadIdx.x & 31;
    const int m    = lane & 15;
    const int hf   = lane >> 4;
    const int wm   = wave >> 1;          // 0..3 -> 32-row strip
    const int wn   = wave & 1;           // 0..1 -> 64-col strip

    // cooperative-load geometry (per thread)
    const int t  = threadIdx.x;
    const int ar = t >> 1, ac = (t & 1) * 16;        // A: 128 rows x 32 cols
    const int br = (t >> 4) << 1;                    // B: k-row pair 0,2,..,30
    const int bc = (t & 15) * 8;                     // B: 8 cols of 128

    const __bf16* Ap  = A + (size_t)(mblk + ar) * K + ac;
    const __bf16* Bp0 = B + (size_t)br * N + nblk + bc;
    const __bf16* Bp1 = B + (size_t)(br + 1) * N + nblk + bc;

    const unsigned ldsA[2] = { lds_addr_of(&As[0][ar][ac]),
                               lds_addr_of(&As[1][ar][ac]) };

    v8f acc[2][4];
    #pragma unroll
    for (int rt = 0; rt < 2; ++rt)
        #pragma unroll
        for (int ct = 0; ct < 4; ++ct)
            acc[rt][ct] = (v8f){0.f,0.f,0.f,0.f,0.f,0.f,0.f,0.f};

    const int nsteps = K >> 5;

    // prologue: stage step 0 (A via async DMA, B via register transpose)
    async_b128_to_lds(ldsA[0],      Ap);
    async_b128_to_lds(ldsA[0] + 16, Ap + 8);
    uint4 b0 = *reinterpret_cast<const uint4*>(Bp0);
    uint4 b1 = *reinterpret_cast<const uint4*>(Bp1);
    {
        const unsigned short* e0 = reinterpret_cast<const unsigned short*>(&b0);
        const unsigned short* e1 = reinterpret_cast<const unsigned short*>(&b1);
        #pragma unroll
        for (int e = 0; e < 8; ++e)
            *reinterpret_cast<unsigned*>(&Bs[0][bc + e][br]) = pack2(e0[e], e1[e]);
    }
    wait_asynccnt0();
    __syncthreads();

    for (int s = 0; s < nsteps; ++s) {
        const int cur = s & 1;
        const int nxt = cur ^ 1;
        const bool more = (s + 1) < nsteps;

        // issue next step's loads early (latency hidden by WMMAs)
        if (more) {
            const size_t ka = (size_t)(s + 1) << 5;
            async_b128_to_lds(ldsA[nxt],      Ap + ka);
            async_b128_to_lds(ldsA[nxt] + 16, Ap + ka + 8);
            b0 = *reinterpret_cast<const uint4*>(Bp0 + ka * N);
            b1 = *reinterpret_cast<const uint4*>(Bp1 + ka * N);
        }

        // compute from current buffer
        Frag af[2];
        #pragma unroll
        for (int rt = 0; rt < 2; ++rt)
            #pragma unroll
            for (int i = 0; i < 8; ++i)
                af[rt].u[i] = *reinterpret_cast<const unsigned int*>(
                    &As[cur][wm * 32 + rt * 16 + m][k0map(i, hf)]);

        #pragma unroll
        for (int ct = 0; ct < 4; ++ct) {
            Frag bf;
            #pragma unroll
            for (int i = 0; i < 8; ++i)
                bf.u[i] = *reinterpret_cast<const unsigned int*>(
                    &Bs[cur][wn * 64 + ct * 16 + m][k0map(i, hf)]);
            #pragma unroll
            for (int rt = 0; rt < 2; ++rt)
                acc[rt][ct] = wmma_bf16(af[rt].v, bf.v, acc[rt][ct]);
        }

        // stage next B buffer (disjoint from the one peers are reading)
        if (more) {
            const unsigned short* e0 = reinterpret_cast<const unsigned short*>(&b0);
            const unsigned short* e1 = reinterpret_cast<const unsigned short*>(&b1);
            #pragma unroll
            for (int e = 0; e < 8; ++e)
                *reinterpret_cast<unsigned*>(&Bs[nxt][bc + e][br]) = pack2(e0[e], e1[e]);
        }
        wait_asynccnt0();
        __syncthreads();
    }

    // epilogue: C/D layout -> row = r + 8*half, col = lane&15
    #pragma unroll
    for (int rt = 0; rt < 2; ++rt)
        #pragma unroll
        for (int ct = 0; ct < 4; ++ct)
            #pragma unroll
            for (int r = 0; r < 8; ++r) {
                int row = mblk + wm * 32 + rt * 16 + r + 8 * hf;
                int col = nblk + wn * 64 + ct * 16 + m;
                C[(size_t)row * N + col] = acc[rt][ct][r];
            }
}

// ---------------------------------------------------------------------------
// RoPE + scatter to head-major bf16 Q/K/V
// ---------------------------------------------------------------------------
__global__ void rope_scatter(const float* __restrict__ QKV,
                             const float* __restrict__ fcos,
                             const float* __restrict__ fsin,
                             __bf16* __restrict__ Q,
                             __bf16* __restrict__ K,
                             __bf16* __restrict__ V) {
    const int row = blockIdx.x;            // 0 .. ROWS-1
    const int b = row / SEQ;
    const int t = row - b * SEQ;
    const float* src = QKV + (size_t)row * QKVN;
    const float* cr = fcos + (size_t)t * (HDIM / 2);
    const float* sr = fsin + (size_t)t * (HDIM / 2);

    for (int idx = threadIdx.x; idx < NH * (HDIM / 2); idx += blockDim.x) {
        int h = idx >> 6, i = idx & 63, d = i * 2;
        float x0 = src[h * HDIM + d];
        float x1 = src[h * HDIM + d + 1];
        float c = cr[i], s = sr[i];
        size_t o = ((size_t)(b * NH + h) * SEQ + t) * HDIM + d;
        Q[o]     = (__bf16)(x0 * c - x1 * s);
        Q[o + 1] = (__bf16)(x0 * s + x1 * c);
    }
    for (int idx = threadIdx.x; idx < NKVH * (HDIM / 2); idx += blockDim.x) {
        int h = idx >> 6, i = idx & 63, d = i * 2;
        float x0 = src[2048 + h * HDIM + d];
        float x1 = src[2048 + h * HDIM + d + 1];
        float c = cr[i], s = sr[i];
        size_t o = ((size_t)(b * NKVH + h) * SEQ + t) * HDIM + d;
        K[o]     = (__bf16)(x0 * c - x1 * s);
        K[o + 1] = (__bf16)(x0 * s + x1 * c);
    }
    for (int idx = threadIdx.x; idx < NKVH * HDIM; idx += blockDim.x) {
        int h = idx >> 7, d = idx & 127;
        size_t o = ((size_t)(b * NKVH + h) * SEQ + t) * HDIM + d;
        V[o] = (__bf16)src[2560 + h * HDIM + d];
    }
}

// ---------------------------------------------------------------------------
// Flash attention: grid (T/128, H, B), block 256 (8 waves).
// Each wave owns a 16-row q tile (4 A-fragments over HD=128) and a 16x128 f32
// accumulator. Keys processed 32 at a time, double-buffered: K chunk staged by
// async global->LDS DMA, V chunk transposed through registers.
// ---------------------------------------------------------------------------
__global__ void flash_attn(const __bf16* __restrict__ Q,
                           const __bf16* __restrict__ Kg,
                           const __bf16* __restrict__ Vg,
                           __bf16* __restrict__ O) {
    __shared__ __bf16 Ks[2][32][HDIM];   // keys row-major (key, hd)
    __shared__ __bf16 Vt[2][HDIM][36];   // V transposed (hd, key), padded
    __shared__ __bf16 Pb[8][16][36];     // per-wave P relayout scratch

    const int qb  = blockIdx.x * 128;
    const int h   = blockIdx.y;
    const int b   = blockIdx.z;
    const int kvh = h >> 2;              // GROUP = 4
    const int wave = threadIdx.x >> 5;
    const int lane = threadIdx.x & 31;
    const int m  = lane & 15;
    const int hf = lane >> 4;

    const __bf16* Qb = Q + (size_t)(b * NH + h) * SEQ * HDIM;
    const __bf16* Kb = Kg + (size_t)(b * NKVH + kvh) * SEQ * HDIM;
    const __bf16* Vb = Vg + (size_t)(b * NKVH + kvh) * SEQ * HDIM;

    // cooperative-load geometry
    const int t  = threadIdx.x;
    const int kr = t >> 3, kc = (t & 7) * 16;   // K: 32 rows x 128 cols
    const int vr = (t >> 4) << 1;               // V: key-row pair 0,2,..,30
    const int vc = (t & 15) * 8;                // V: 8 hd cols

    const unsigned ldsK[2] = { lds_addr_of(&Ks[0][kr][kc]),
                               lds_addr_of(&Ks[1][kr][kc]) };

    // load Q fragments (rows qb + 16*wave + m)
    Frag qf[4];
    {
        const size_t qrow = (size_t)(qb + wave * 16 + m) * HDIM;
        #pragma unroll
        for (int c = 0; c < 4; ++c)
            #pragma unroll
            for (int i = 0; i < 8; ++i)
                qf[c].u[i] = *reinterpret_cast<const unsigned int*>(
                    &Qb[qrow + c * 32 + k0map(i, hf)]);
    }

    v8f Oacc[8];
    #pragma unroll
    for (int c = 0; c < 8; ++c) Oacc[c] = (v8f){0.f,0.f,0.f,0.f,0.f,0.f,0.f,0.f};
    float mrow[8], lrow[8];
    #pragma unroll
    for (int r = 0; r < 8; ++r) { mrow[r] = -3.0e38f; lrow[r] = 0.f; }

    const float scale = 0.08838834764831845f;  // 1/sqrt(128)
    const int qrow0 = qb + wave * 16 + 8 * hf; // + r gives this lane's rows
    const int nchunks = (qb + 128) >> 5;

    // prologue: stage chunk 0
    uint4 vv0, vv1;
    async_b128_to_lds(ldsK[0],      &Kb[(size_t)kr * HDIM + kc]);
    async_b128_to_lds(ldsK[0] + 16, &Kb[(size_t)kr * HDIM + kc + 8]);
    vv0 = *reinterpret_cast<const uint4*>(&Vb[(size_t)vr * HDIM + vc]);
    vv1 = *reinterpret_cast<const uint4*>(&Vb[(size_t)(vr + 1) * HDIM + vc]);
    {
        const unsigned short* e0 = reinterpret_cast<const unsigned short*>(&vv0);
        const unsigned short* e1 = reinterpret_cast<const unsigned short*>(&vv1);
        #pragma unroll
        for (int e = 0; e < 8; ++e)
            *reinterpret_cast<unsigned*>(&Vt[0][vc + e][vr]) = pack2(e0[e], e1[e]);
    }
    wait_asynccnt0();
    __syncthreads();

    for (int s = 0; s < nchunks; ++s) {
        const int j0 = s << 5;
        const int cur = s & 1;
        const int nxt = cur ^ 1;
        const bool more = (s + 1) < nchunks;

        // issue next chunk's loads early
        if (more) {
            const int j1 = j0 + 32;
            async_b128_to_lds(ldsK[nxt],      &Kb[(size_t)(j1 + kr) * HDIM + kc]);
            async_b128_to_lds(ldsK[nxt] + 16, &Kb[(size_t)(j1 + kr) * HDIM + kc + 8]);
            vv0 = *reinterpret_cast<const uint4*>(&Vb[(size_t)(j1 + vr) * HDIM + vc]);
            vv1 = *reinterpret_cast<const uint4*>(&Vb[(size_t)(j1 + vr + 1) * HDIM + vc]);
        }

        // S = Q * K^T : two 16x16 column blocks over 32 keys
        v8f s0 = (v8f){0.f,0.f,0.f,0.f,0.f,0.f,0.f,0.f};
        v8f s1 = (v8f){0.f,0.f,0.f,0.f,0.f,0.f,0.f,0.f};
        #pragma unroll
        for (int c = 0; c < 4; ++c) {
            Frag kf;
            #pragma unroll
            for (int i = 0; i < 8; ++i)
                kf.u[i] = *reinterpret_cast<const unsigned int*>(
                    &Ks[cur][m][c * 32 + k0map(i, hf)]);
            s0 = wmma_bf16(qf[c].v, kf.v, s0);
        }
        #pragma unroll
        for (int c = 0; c < 4; ++c) {
            Frag kf;
            #pragma unroll
            for (int i = 0; i < 8; ++i)
                kf.u[i] = *reinterpret_cast<const unsigned int*>(
                    &Ks[cur][16 + m][c * 32 + k0map(i, hf)]);
            s1 = wmma_bf16(qf[c].v, kf.v, s1);
        }

        // causal mask + online softmax (row = r + 8*hf, col n = lane&15)
        const int key0 = j0 + m;
        const int key1 = j0 + 16 + m;
        #pragma unroll
        for (int r = 0; r < 8; ++r) {
            const int qrow = qrow0 + r;
            float v0 = s0[r] * scale;
            float v1 = s1[r] * scale;
            if (key0 > qrow) v0 = -1.0e30f;
            if (key1 > qrow) v1 = -1.0e30f;
            float rm = fmaxf(v0, v1);
            #pragma unroll
            for (int d = 1; d < 16; d <<= 1)
                rm = fmaxf(rm, __shfl_xor(rm, d, 32));
            const float mnew = fmaxf(mrow[r], rm);
            const float corr = __expf(mrow[r] - mnew);
            const float p0 = __expf(v0 - mnew);
            const float p1 = __expf(v1 - mnew);
            float rs = p0 + p1;
            #pragma unroll
            for (int d = 1; d < 16; d <<= 1)
                rs += __shfl_xor(rs, d, 32);
            lrow[r] = lrow[r] * corr + rs;
            mrow[r] = mnew;
            #pragma unroll
            for (int c = 0; c < 8; ++c) Oacc[c][r] *= corr;
            Pb[wave][r + 8 * hf][m]      = (__bf16)p0;
            Pb[wave][r + 8 * hf][16 + m] = (__bf16)p1;
        }

        // re-read P as an A fragment (same-wave LDS RAW: in-order DS ops)
        Frag pf;
        #pragma unroll
        for (int i = 0; i < 8; ++i)
            pf.u[i] = *reinterpret_cast<const unsigned int*>(
                &Pb[wave][m][k0map(i, hf)]);

        // O += P (16x32) * V (32x128)
        #pragma unroll
        for (int c = 0; c < 8; ++c) {
            Frag vf;
            #pragma unroll
            for (int i = 0; i < 8; ++i)
                vf.u[i] = *reinterpret_cast<const unsigned int*>(
                    &Vt[cur][c * 16 + m][k0map(i, hf)]);
            Oacc[c] = wmma_bf16(pf.v, vf.v, Oacc[c]);
        }

        // stage next V buffer (disjoint from the one peers are reading)
        if (more) {
            const unsigned short* e0 = reinterpret_cast<const unsigned short*>(&vv0);
            const unsigned short* e1 = reinterpret_cast<const unsigned short*>(&vv1);
            #pragma unroll
            for (int e = 0; e < 8; ++e)
                *reinterpret_cast<unsigned*>(&Vt[nxt][vc + e][vr]) = pack2(e0[e], e1[e]);
        }
        wait_asynccnt0();
        __syncthreads();
    }

    // normalize and store O as [B,T,D] bf16 with channel = h*HD + hd
    #pragma unroll
    for (int r = 0; r < 8; ++r) {
        const float inv = 1.0f / lrow[r];
        const int qrow = qrow0 + r;
        const size_t base = ((size_t)b * SEQ + qrow) * DIM + h * HDIM;
        #pragma unroll
        for (int c = 0; c < 8; ++c)
            O[base + c * 16 + m] = (__bf16)(Oacc[c][r] * inv);
    }
}

// ---------------------------------------------------------------------------
// Launch
// ---------------------------------------------------------------------------
extern "C" void kernel_launch(void* const* d_in, const int* in_sizes, int n_in,
                              void* d_out, int out_size, void* d_ws, size_t ws_size,
                              hipStream_t stream) {
    (void)in_sizes; (void)n_in; (void)out_size; (void)ws_size;

    const float* x    = (const float*)d_in[0];
    const float* fcos = (const float*)d_in[1];
    const float* fsin = (const float*)d_in[2];
    const float* Wq   = (const float*)d_in[3];
    const float* Wk   = (const float*)d_in[4];
    const float* Wv   = (const float*)d_in[5];
    const float* Wo   = (const float*)d_in[6];
    float* out = (float*)d_out;

    char* ws = (char*)d_ws;
    const size_t SZ_XB   = (size_t)ROWS * DIM * 2;          // bf16 x; reused for O
    const size_t SZ_WQKV = (size_t)DIM * QKVN * 2;          // reused for Wo bf16
    const size_t SZ_QKV  = (size_t)ROWS * QKVN * 4;         // fp32 QKV
    const size_t SZ_Q    = (size_t)BATCH * NH * SEQ * HDIM * 2;
    const size_t SZ_K    = (size_t)BATCH * NKVH * SEQ * HDIM * 2;

    __bf16* xb    = (__bf16*)(ws);
    __bf16* Wqkvb = (__bf16*)(ws + SZ_XB);
    float*  QKV   = (float*) (ws + SZ_XB + SZ_WQKV);
    __bf16* Qb    = (__bf16*)(ws + SZ_XB + SZ_WQKV + SZ_QKV);
    __bf16* Kb    = (__bf16*)(ws + SZ_XB + SZ_WQKV + SZ_QKV + SZ_Q);
    __bf16* Vb    = (__bf16*)(ws + SZ_XB + SZ_WQKV + SZ_QKV + SZ_Q + SZ_K);
    __bf16* Ob    = xb;      // reuse: x bf16 dead after QKV GEMM
    __bf16* Wob   = Wqkvb;   // reuse: Wqkv bf16 dead after QKV GEMM

    // 1) converts
    cvt_f32_to_bf16<<<4096, 256, 0, stream>>>(x, xb, ROWS * DIM);
    pack_wqkv_bf16<<<4096, 256, 0, stream>>>(Wq, Wk, Wv, Wqkvb);

    // 2) QKV = x @ [Wq|Wk|Wv]   (4096 x 3072 x 2048)
    gemm_bf16_f32<<<dim3(QKVN / 128, ROWS / 128), 256, 0, stream>>>(
        xb, Wqkvb, QKV, ROWS, QKVN, DIM);

    // Wo -> bf16 (after gemm1 so the Wqkv region can be reused)
    cvt_f32_to_bf16<<<4096, 256, 0, stream>>>(Wo, Wob, DIM * DIM);

    // 3) RoPE + head-major scatter
    rope_scatter<<<ROWS, 256, 0, stream>>>(QKV, fcos, fsin, Qb, Kb, Vb);

    // 4) flash attention
    flash_attn<<<dim3(SEQ / 128, NH, BATCH), 256, 0, stream>>>(Qb, Kb, Vb, Ob);

    // 5) y = O @ Wo  (4096 x 2048 x 2048) -> d_out fp32
    gemm_bf16_f32<<<dim3(DIM / 128, ROWS / 128), 256, 0, stream>>>(
        Ob, Wob, out, ROWS, DIM, DIM);
}
